// PrunableMixtralSparseMoeBlockWrapper_59949153518005
// MI455X (gfx1250) — matrix operand order repaired and checked by reading, not compiled, not run on hardware
//
#include <hip/hip_runtime.h>
#include <cstddef>

typedef __attribute__((ext_vector_type(16))) __bf16 v16bf;
typedef __attribute__((ext_vector_type(8)))  __bf16 v8bf;
typedef __attribute__((ext_vector_type(8)))  float  v8f;
typedef __attribute__((ext_vector_type(4)))  unsigned int u32x4;
typedef __attribute__((ext_vector_type(8)))  int i32x8;
typedef __attribute__((ext_vector_type(4)))  int i32x4;

namespace {
constexpr int kT = 8192;   // B*S tokens
constexpr int kH = 2048;   // hidden
constexpr int kF = 4096;   // intermediate
constexpr int kE = 8;      // experts
constexpr int BM = 128;    // block tile M (token slots)
constexpr int BN = 64;     // block tile N
constexpr int BK = 32;     // K chunk (one WMMA K)
}

#if defined(__HIP_DEVICE_COMPILE__) && __has_builtin(__builtin_amdgcn_tensor_load_to_lds)
#define HAVE_TDM 1
#else
#define HAVE_TDM 0
#endif

#if defined(__HIP_DEVICE_COMPILE__) && __has_builtin(__builtin_amdgcn_s_wait_tensorcnt)
#define WAIT_TENSOR() __builtin_amdgcn_s_wait_tensorcnt((short)0)
#elif defined(__HIP_DEVICE_COMPILE__)
#define WAIT_TENSOR() asm volatile("s_wait_tensorcnt 0x0" ::: "memory")
#else
#define WAIT_TENSOR()
#endif

__device__ __forceinline__ v8bf cvt8(const float4 a, const float4 b) {
  v8bf r;
  r[0] = (__bf16)a.x; r[1] = (__bf16)a.y; r[2] = (__bf16)a.z; r[3] = (__bf16)a.w;
  r[4] = (__bf16)b.x; r[5] = (__bf16)b.y; r[6] = (__bf16)b.z; r[7] = (__bf16)b.w;
  return r;
}

#if HAVE_TDM
// Issue one TDM 2D tile load: BM rows x BK cols of bf16, row stride kF elems.
// D# packing per CDNA5 ISA.
// 6-arg builtin form: (g0, g1, g2, g3, g_extra, cpol)
__device__ __forceinline__ void tdm_load_tileA(unsigned long long gaddr,
                                               unsigned lds_addr) {
  u32x4 g0;
  g0[0] = 1u;                                                        // count=1
  g0[1] = lds_addr;                                                  // lds_addr
  g0[2] = (unsigned)(gaddr & 0xFFFFFFFFull);                         // gaddr lo
  g0[3] = (unsigned)((gaddr >> 32) & 0x01FFFFFFull) | 0x80000000u;   // gaddr hi | type=2
  i32x8 g1;
  g1[0] = 1 << 16;                                                   // data_size=1 (2 bytes)
  g1[1] = (int)(((unsigned)kF & 0xFFFFu) << 16);                     // tensor_dim0[15:0]
  g1[2] = (int)(((unsigned)kF >> 16) & 0xFFFFu)
        | (int)(((unsigned)kT & 0xFFFFu) << 16);                     // dim0[31:16] | dim1[15:0]
  g1[3] = (int)(((unsigned)kT >> 16) & 0xFFFFu) | (BK << 16);        // dim1[31:16] | tile_dim0
  g1[4] = BM;                                                        // tile_dim1 | tile_dim2=0
  g1[5] = kF;                                                        // dim0_stride[31:0]
  g1[6] = 0;
  g1[7] = 0;
  i32x4 z4 = {};
  i32x8 z8 = {};
  __builtin_amdgcn_tensor_load_to_lds(g0, g1, z4, z4, z8, 0);
}
#endif

// ------- Router: logits -> softmax -> top2 -> combine + per-expert lists -----
__global__ __launch_bounds__(256) void moe_router(
    const float* __restrict__ x, const float* __restrict__ gate_w,
    float* __restrict__ logits_out, float* __restrict__ combine,
    int* __restrict__ counts, int* __restrict__ idxlist) {
  const int lane  = threadIdx.x & 31;
  const int token = blockIdx.x * 8 + (threadIdx.x >> 5);  // 8 waves/block
  const float* xr = x + (size_t)token * kH;

  float acc[kE];
#pragma unroll
  for (int e = 0; e < kE; ++e) acc[e] = 0.f;
  for (int k = lane; k < kH; k += 32) {
    const float xv = xr[k];
#pragma unroll
    for (int e = 0; e < kE; ++e) acc[e] = fmaf(xv, gate_w[e * kH + k], acc[e]);
  }
#pragma unroll
  for (int e = 0; e < kE; ++e) {
#pragma unroll
    for (int off = 16; off > 0; off >>= 1) acc[e] += __shfl_xor(acc[e], off, 32);
  }
  if (lane == 0) {
    float mx = acc[0];
#pragma unroll
    for (int e = 1; e < kE; ++e) mx = fmaxf(mx, acc[e]);
    float p[kE];
#pragma unroll
    for (int e = 0; e < kE; ++e) p[e] = __expf(acc[e] - mx);
    int i0 = 0;
#pragma unroll
    for (int e = 1; e < kE; ++e) if (p[e] > p[i0]) i0 = e;
    int i1 = (i0 == 0) ? 1 : 0;
#pragma unroll
    for (int e = 0; e < kE; ++e) if (e != i0 && p[e] > p[i1]) i1 = e;
    const float inv = 1.f / (p[i0] + p[i1]);
#pragma unroll
    for (int e = 0; e < kE; ++e) {
      logits_out[(size_t)token * kE + e] = acc[e];
      combine[(size_t)token * kE + e] =
          (e == i0) ? p[i0] * inv : ((e == i1) ? p[i1] * inv : 0.f);
    }
    const int s0 = atomicAdd(&counts[i0], 1);
    idxlist[(size_t)i0 * kT + s0] = token;
    const int s1 = atomicAdd(&counts[i1], 1);
    idxlist[(size_t)i1 * kT + s1] = token;
  }
}

// ------- GEMM1 (gathered): h[slot] = silu(x[tok] w1^T) * (x[tok] w3^T) -------
__global__ __launch_bounds__(256) void moe_gemm1(
    const float* __restrict__ x, const float* __restrict__ w1,
    const float* __restrict__ w3, __bf16* __restrict__ hbuf,
    const int* __restrict__ counts, const int* __restrict__ idxlist, int e) {
  const int cnt = counts[e];
  const int bm = blockIdx.x * BM;
  if (bm >= cnt) return;                     // sparse dispatch: skip empty tiles
  const int bn = blockIdx.y * BN;

  __shared__ __bf16 As[2][BM][BK];
  __shared__ __bf16 B1s[2][BN][BK];
  __shared__ __bf16 B3s[2][BN][BK];

  const int lane  = threadIdx.x & 31;
  const int wave  = threadIdx.x >> 5;
  const int wm    = (wave >> 1) * 32;        // 0,32,64,96
  const int wn    = (wave & 1) * 32;         // 0,32
  const int lrow  = lane & 15;
  const int lhalf = lane >> 4;

  const int lar = threadIdx.x >> 1;          // A loader: 2 thr/row, 16 f32 each
  const int lac = (threadIdx.x & 1) * 16;
  const int lbr = threadIdx.x >> 2;          // B loader: 4 thr/row, 8 f32 each
  const int lbc = (threadIdx.x & 3) * 8;

  const int slot = (bm + lar < cnt) ? (bm + lar) : (cnt - 1);
  const int tok  = idxlist[(size_t)e * kT + slot];
  const float* gA = x  + (size_t)tok * kH + lac;
  const float* g1 = w1 + ((size_t)e * kF + bn + lbr) * kH + lbc;
  const float* g3 = w3 + ((size_t)e * kF + bn + lbr) * kH + lbc;

  v8f zero = {};
  v8f acc1[2][2], acc3[2][2];
#pragma unroll
  for (int tm = 0; tm < 2; ++tm)
#pragma unroll
    for (int tn = 0; tn < 2; ++tn) { acc1[tm][tn] = zero; acc3[tm][tn] = zero; }

  constexpr int NK = kH / BK;
  float4 ra0, ra1, ra2, ra3, rc0, rc1, rd0, rd1;
  {  // prologue: chunk 0 -> regs -> buf 0
    const float4* pa = (const float4*)gA;
    ra0 = pa[0]; ra1 = pa[1]; ra2 = pa[2]; ra3 = pa[3];
    const float4* p1 = (const float4*)g1; rc0 = p1[0]; rc1 = p1[1];
    const float4* p3 = (const float4*)g3; rd0 = p3[0]; rd1 = p3[1];
  }
  *(v8bf*)&As[0][lar][lac]     = cvt8(ra0, ra1);
  *(v8bf*)&As[0][lar][lac + 8] = cvt8(ra2, ra3);
  *(v8bf*)&B1s[0][lbr][lbc]    = cvt8(rc0, rc1);
  *(v8bf*)&B3s[0][lbr][lbc]    = cvt8(rd0, rd1);

  for (int kc = 0; kc < NK; ++kc) {
    const int buf = kc & 1;
    if (kc + 1 < NK) {  // stage next chunk while this one computes
      const int k0 = (kc + 1) * BK;
      const float4* pa = (const float4*)(gA + k0);
      ra0 = pa[0]; ra1 = pa[1]; ra2 = pa[2]; ra3 = pa[3];
      const float4* p1 = (const float4*)(g1 + k0); rc0 = p1[0]; rc1 = p1[1];
      const float4* p3 = (const float4*)(g3 + k0); rd0 = p3[0]; rd1 = p3[1];
      __builtin_prefetch(g1 + k0 + BK, 0, 1);
      __builtin_prefetch(g3 + k0 + BK, 0, 1);
    }
    __syncthreads();  // publish buf; prior reads of buf^1 done

    const int ka = lhalf * 8;
    const int kb = lhalf * 8;
    v16bf af[2], b1f[2], b3f[2];
#pragma unroll
    for (int tm = 0; tm < 2; ++tm) {
      v8bf* ap = (v8bf*)&af[tm];
      ap[0] = *(const v8bf*)&As[buf][wm + tm * 16 + lrow][ka];
      ap[1] = *(const v8bf*)&As[buf][wm + tm * 16 + lrow][ka + 16];
    }
#pragma unroll
    for (int tn = 0; tn < 2; ++tn) {
      v8bf* bp = (v8bf*)&b1f[tn];
      bp[0] = *(const v8bf*)&B1s[buf][wn + tn * 16 + lrow][kb];
      bp[1] = *(const v8bf*)&B1s[buf][wn + tn * 16 + lrow][kb + 16];
      v8bf* cp = (v8bf*)&b3f[tn];
      cp[0] = *(const v8bf*)&B3s[buf][wn + tn * 16 + lrow][kb];
      cp[1] = *(const v8bf*)&B3s[buf][wn + tn * 16 + lrow][kb + 16];
    }
#pragma unroll
    for (int tm = 0; tm < 2; ++tm)
#pragma unroll
      for (int tn = 0; tn < 2; ++tn) {
        acc1[tm][tn] = __builtin_amdgcn_wmma_f32_16x16x32_bf16(
            false, af[tm], false, b1f[tn], (short)0, acc1[tm][tn], false, false);
        acc3[tm][tn] = __builtin_amdgcn_wmma_f32_16x16x32_bf16(
            false, af[tm], false, b3f[tn], (short)0, acc3[tm][tn], false, false);
      }

    if (kc + 1 < NK) {
      const int nb = buf ^ 1;
      *(v8bf*)&As[nb][lar][lac]     = cvt8(ra0, ra1);
      *(v8bf*)&As[nb][lar][lac + 8] = cvt8(ra2, ra3);
      *(v8bf*)&B1s[nb][lbr][lbc]    = cvt8(rc0, rc1);
      *(v8bf*)&B3s[nb][lbr][lbc]    = cvt8(rd0, rd1);
    }
  }

  // Epilogue: SwiGLU (fast rcp sigmoid), store bf16 h at compacted slots
#pragma unroll
  for (int tm = 0; tm < 2; ++tm)
#pragma unroll
    for (int r = 0; r < 8; ++r) {
      const int gm = bm + wm + tm * 16 + lhalf * 8 + r;
      if (gm < cnt) {
#pragma unroll
        for (int tn = 0; tn < 2; ++tn) {
          const int gf = bn + wn + tn * 16 + lrow;
          const float v1 = acc1[tm][tn][r];
          const float v3 = acc3[tm][tn][r];
          const float sig = __builtin_amdgcn_rcpf(1.f + __expf(-v1));
          hbuf[(size_t)gm * kF + gf] = (__bf16)(v1 * sig * v3);
        }
      }
    }
}

// ------- GEMM2 (scatter): out[tok] += (h[slot] w2^T) * combine[tok,e] --------
__global__ __launch_bounds__(256) void moe_gemm2(
    const __bf16* __restrict__ hbuf, const float* __restrict__ w2,
    const float* __restrict__ combine, const int* __restrict__ counts,
    const int* __restrict__ idxlist, float* __restrict__ out, int e) {
  const int cnt = counts[e];
  const int bm = blockIdx.x * BM;
  if (bm >= cnt) return;
  const int bn = blockIdx.y * BN;

  __shared__ __bf16 As[2][BM][BK];
  __shared__ __bf16 Bs[2][BN][BK];

  const int lane  = threadIdx.x & 31;
  const int wave  = threadIdx.x >> 5;
  const int wm    = (wave >> 1) * 32;
  const int wn    = (wave & 1) * 32;
  const int lrow  = lane & 15;
  const int lhalf = lane >> 4;

  const int lar = threadIdx.x >> 1;
  const int lac = (threadIdx.x & 1) * 16;
  const int lbr = threadIdx.x >> 2;
  const int lbc = (threadIdx.x & 3) * 8;

  const __bf16* gA = hbuf + (size_t)(bm + lar) * kF + lac;   // fallback path
  const float*  gB = w2 + ((size_t)e * kH + bn + lbr) * kF + lbc;

  v8f zero = {};
  v8f acc[2][2];
#pragma unroll
  for (int tm = 0; tm < 2; ++tm)
#pragma unroll
    for (int tn = 0; tn < 2; ++tn) acc[tm][tn] = zero;

  constexpr int NK = kF / BK;
#if HAVE_TDM
  const unsigned long long habase =
      (unsigned long long)(const void*)(hbuf + (size_t)bm * kF);
  const unsigned ldsA0 = (unsigned)(unsigned long long)(const void*)&As[0][0][0];
  constexpr unsigned ABYTES = BM * BK * 2;
  if (wave == 0) tdm_load_tileA(habase, ldsA0);   // chunk 0 -> As[0] via TDM
#else
  v8bf va0, va1;
  { const v8bf* pa = (const v8bf*)gA; va0 = pa[0]; va1 = pa[1]; }
  *(v8bf*)&As[0][lar][lac]     = va0;
  *(v8bf*)&As[0][lar][lac + 8] = va1;
#endif
  float4 rb0, rb1;
  { const float4* pb = (const float4*)gB; rb0 = pb[0]; rb1 = pb[1]; }
  *(v8bf*)&Bs[0][lbr][lbc] = cvt8(rb0, rb1);

  for (int kc = 0; kc < NK; ++kc) {
    const int buf = kc & 1;
    if (kc + 1 < NK) {
      const int k0 = (kc + 1) * BK;
      const float4* pb = (const float4*)(gB + k0);
      rb0 = pb[0]; rb1 = pb[1];
      __builtin_prefetch(gB + k0 + BK, 0, 1);
#if !HAVE_TDM
      const v8bf* pa = (const v8bf*)(gA + k0);
      va0 = pa[0]; va1 = pa[1];
#endif
    }
#if HAVE_TDM
    WAIT_TENSOR();      // wave 0: chunk kc's DMA landed; others fall through
#endif
    __syncthreads();
#if HAVE_TDM
    if (kc + 1 < NK && wave == 0)  // overlap next tile's DMA with this compute
      tdm_load_tileA(habase + (unsigned long long)(kc + 1) * BK * 2,
                     ldsA0 + (unsigned)((buf ^ 1) * ABYTES));
#endif

    const int ka = lhalf * 8;
    const int kb = lhalf * 8;
    v16bf af[2], bfr[2];
#pragma unroll
    for (int tm = 0; tm < 2; ++tm) {
      v8bf* ap = (v8bf*)&af[tm];
      ap[0] = *(const v8bf*)&As[buf][wm + tm * 16 + lrow][ka];
      ap[1] = *(const v8bf*)&As[buf][wm + tm * 16 + lrow][ka + 16];
    }
#pragma unroll
    for (int tn = 0; tn < 2; ++tn) {
      v8bf* bp = (v8bf*)&bfr[tn];
      bp[0] = *(const v8bf*)&Bs[buf][wn + tn * 16 + lrow][kb];
      bp[1] = *(const v8bf*)&Bs[buf][wn + tn * 16 + lrow][kb + 16];
    }
#pragma unroll
    for (int tm = 0; tm < 2; ++tm)
#pragma unroll
      for (int tn = 0; tn < 2; ++tn)
        acc[tm][tn] = __builtin_amdgcn_wmma_f32_16x16x32_bf16(
            false, af[tm], false, bfr[tn], (short)0, acc[tm][tn], false, false);

    if (kc + 1 < NK) {
      const int nb = buf ^ 1;
      *(v8bf*)&Bs[nb][lbr][lbc] = cvt8(rb0, rb1);
#if !HAVE_TDM
      *(v8bf*)&As[nb][lar][lac]     = va0;
      *(v8bf*)&As[nb][lar][lac + 8] = va1;
#endif
    }
  }

  // Epilogue: scatter back to tokens (experts serialized by stream order)
#pragma unroll
  for (int tm = 0; tm < 2; ++tm)
#pragma unroll
    for (int r = 0; r < 8; ++r) {
      const int gm = bm + wm + tm * 16 + lhalf * 8 + r;
      if (gm < cnt) {
        const int tok = idxlist[(size_t)e * kT + gm];
        const float cw = combine[(size_t)tok * kE + e];
#pragma unroll
        for (int tn = 0; tn < 2; ++tn) {
          const int gn = bn + wn + tn * 16 + lrow;
          const size_t idx = (size_t)tok * kH + gn;
          out[idx] += acc[tm][tn][r] * cw;
        }
      }
    }
}

extern "C" void kernel_launch(void* const* d_in, const int* in_sizes, int n_in,
                              void* d_out, int out_size, void* d_ws, size_t ws_size,
                              hipStream_t stream) {
  const float* x      = (const float*)d_in[0];
  const float* gate_w = (const float*)d_in[1];
  const float* w1     = (const float*)d_in[2];
  const float* w3     = (const float*)d_in[3];
  const float* w2     = (const float*)d_in[4];

  float* out    = (float*)d_out;
  float* logits = out + (size_t)kT * kH;                 // d_out tail: [T,E]

  // workspace layout
  char* ws = (char*)d_ws;
  int*    counts  = (int*)ws;                                        // kE ints
  float*  combine = (float*)(ws + 256);                              // [T,E] f32
  int*    idxlist = (int*)(ws + 256 + (size_t)kT * kE * 4);          // [E,T] i32
  __bf16* hbuf    = (__bf16*)(ws + 256 + (size_t)kT * kE * 8);       // [T,F] bf16

  (void)hipMemsetAsync(out, 0, (size_t)kT * kH * sizeof(float), stream);
  (void)hipMemsetAsync(counts, 0, kE * sizeof(int), stream);
  moe_router<<<kT / 8, 256, 0, stream>>>(x, gate_w, logits, combine, counts, idxlist);
  for (int e = 0; e < kE; ++e) {
    moe_gemm1<<<dim3(kT / BM, kF / BN), 256, 0, stream>>>(
        x, w1, w3, hbuf, counts, idxlist, e);
    moe_gemm2<<<dim3(kT / BM, kH / BN), 256, 0, stream>>>(
        hbuf, w2, combine, counts, idxlist, out, e);
  }
}
